// FoldingNetEncoder_10934986735971
// MI455X (gfx1250) — matrix operand order repaired and checked by reading, not compiled, not run on hardware
//
#include <hip/hip_runtime.h>

#define B_  8
#define N_  2048
#define KNN 16
#define BN  (B_ * N_)

typedef __attribute__((ext_vector_type(16))) __bf16         v16bf;
typedef __attribute__((ext_vector_type(8)))  float          v8f;
typedef __attribute__((ext_vector_type(8)))  unsigned short ushort8;

// ---------- helpers ----------
static __device__ __forceinline__ unsigned short f2bf(float f) {
    unsigned u = __builtin_bit_cast(unsigned, f);
    unsigned r = (u + 0x7fffu + ((u >> 16) & 1u)) >> 16;   // RNE
    return (unsigned short)r;
}
static __device__ __forceinline__ unsigned fkey(float f) {  // monotone float->uint
    int s = __builtin_bit_cast(int, f);
    return (s >= 0) ? ((unsigned)s | 0x80000000u) : (unsigned)(~s);
}
static __device__ __forceinline__ float fdec(unsigned u) {
    int s = (u & 0x80000000u) ? (int)(u & 0x7fffffffu) : ~(int)u;
    return __builtin_bit_cast(float, s);
}

// A fragment: 16x32 bf16, M = mBase + (lane&15); per ISA table the 16 halfwords
// per lane are K = {0..7,16..23} (+8 for lanes 16..31). Row-major (rows, ld) src.
static __device__ __forceinline__ v16bf load_a_frag(const unsigned short* base, int ld,
                                                    int mBase, int kk) {
    int lane = threadIdx.x & 31;
    const unsigned short* p = base + (size_t)(mBase + (lane & 15)) * ld + kk + ((lane >> 4) << 3);
    ushort8 c0 = *(const ushort8*)(p);
    ushort8 c1 = *(const ushort8*)(p + 16);
    v16bf a;
#pragma unroll
    for (int j = 0; j < 8; ++j) {
        a[j]     = __builtin_bit_cast(__bf16, (unsigned short)c0[j]);
        a[j + 8] = __builtin_bit_cast(__bf16, (unsigned short)c1[j]);
    }
    return a;
}
// B fragment: 32x16 bf16, col N = nBase + (lane&15); K = {0..7,16..23} per lane,
// +8 for lanes 16..31. Source row-major (cols, ld) i.e. col n of B == row n of src.
static __device__ __forceinline__ v16bf load_b_frag(const unsigned short* base, int ld,
                                                    int nBase, int kk) {
    int lane = threadIdx.x & 31;
    const unsigned short* p = base + (size_t)(nBase + (lane & 15)) * ld + kk + ((lane >> 4) << 3);
    ushort8 c0 = *(const ushort8*)(p);
    ushort8 c1 = *(const ushort8*)(p + 16);
    v16bf b;
#pragma unroll
    for (int j = 0; j < 8; ++j) {
        b[j]     = __builtin_bit_cast(__bf16, (unsigned short)c0[j]);
        b[j + 8] = __builtin_bit_cast(__bf16, (unsigned short)c1[j]);
    }
    return b;
}

// ---------- conversion / small kernels ----------
__global__ void k_convert(const float* __restrict__ src, unsigned short* __restrict__ dst,
                          int ldSrc, int Creal, int Kpad, int total) {
    int id = blockIdx.x * blockDim.x + threadIdx.x;
    if (id >= total) return;
    int m = id / Kpad, k = id - m * Kpad;
    dst[id] = (k < Creal) ? f2bf(src[(size_t)m * ldSrc + k]) : (unsigned short)0;
}

__global__ void k_norms(const float* __restrict__ src, int ldSrc, int Creal,
                        float* __restrict__ out) {
    int id = blockIdx.x * blockDim.x + threadIdx.x;
    if (id >= BN) return;
    float s = 0.f;
    for (int c = 0; c < Creal; ++c) {
        float v = src[(size_t)id * ldSrc + c];
        s += v * v;
    }
    out[id] = s;
}

__global__ void k_zero(unsigned* __restrict__ p, int n) {
    int id = blockIdx.x * blockDim.x + threadIdx.x;
    if (id < n) p[id] = 0u;
}

__global__ void k_cov(const float* __restrict__ pts, const int* __restrict__ idx,
                      float* __restrict__ out) {
    int id = blockIdx.x * blockDim.x + threadIdx.x;
    if (id >= BN) return;
    int b = id / N_, n = id - b * N_;
    const float* P = pts + (size_t)b * N_ * 3;
    int i0 = idx[id * KNN + 0], i1 = idx[id * KNN + 1];
    float a0 = P[i0 * 3], a1 = P[i0 * 3 + 1], a2 = P[i0 * 3 + 2];
    float c0 = P[i1 * 3], c1 = P[i1 * 3 + 1], c2 = P[i1 * 3 + 2];
    float* o = out + (size_t)id * 32;
    o[0] = P[n * 3]; o[1] = P[n * 3 + 1]; o[2] = P[n * 3 + 2];
    o[3] = a0 * c0; o[4]  = a0 * c1; o[5]  = a0 * c2;
    o[6] = a1 * c0; o[7]  = a1 * c1; o[8]  = a1 * c2;
    o[9] = a2 * c0; o[10] = a2 * c1; o[11] = a2 * c2;
    for (int c = 12; c < 32; ++c) o[c] = 0.f;
}

__global__ void k_maxpool(const float* __restrict__ in, const int* __restrict__ idx,
                          float* __restrict__ out, int C, int ld) {
    int id = blockIdx.x * blockDim.x + threadIdx.x;
    if (id >= BN * C) return;
    int m = id / C, c = id - m * C;
    int b = m / N_;
    const int* ii = idx + m * KNN;
    float mx = -3.4e38f;
#pragma unroll
    for (int j = 0; j < KNN; ++j) {
        int nb = ii[j];
        mx = fmaxf(mx, in[(size_t)(b * N_ + nb) * ld + c]);
    }
    out[(size_t)m * ld + c] = mx;
}

// ---------- WMMA GEMM: out[m,n] = relu?(sum_k A[m,k]*W[n,k] + bias[n]) ----------
// Block = 128 threads (4 waves). Each wave owns a 32x64 output tile:
// 2 A-frags + 4 B-frags -> 8 WMMA per K-step (0.75 fragment loads per WMMA).
__global__ __launch_bounds__(128) void k_gemm(const unsigned short* __restrict__ A,
                                              const unsigned short* __restrict__ W,
                                              const float* __restrict__ bias,
                                              float* __restrict__ out,
                                              int K, int ldA, int ldOut, int relu) {
    int wave = threadIdx.x >> 5;
    int nTile = blockIdx.x << 6;                  // 64 output channels per block
    int mW = (blockIdx.y << 7) + (wave << 5);     // 32 rows per wave
    v8f acc[2][4] = {};
    for (int kk = 0; kk < K; kk += 32) {
        v16bf a0 = load_a_frag(A, ldA, mW, kk);
        v16bf a1 = load_a_frag(A, ldA, mW + 16, kk);
#pragma unroll
        for (int nt = 0; nt < 4; ++nt) {
            v16bf b = load_b_frag(W, K, nTile + (nt << 4), kk);
            acc[0][nt] = __builtin_amdgcn_wmma_f32_16x16x32_bf16(false, a0, false, b,
                                                                 (short)0, acc[0][nt], false, false);
            acc[1][nt] = __builtin_amdgcn_wmma_f32_16x16x32_bf16(false, a1, false, b,
                                                                 (short)0, acc[1][nt], false, false);
        }
    }
    int lane = threadIdx.x & 31;
    int hi = (lane >> 4) << 3;
#pragma unroll
    for (int nt = 0; nt < 4; ++nt) {
        int col = nTile + (nt << 4) + (lane & 15);
        float bv = bias ? bias[col] : 0.f;
#pragma unroll
        for (int mt = 0; mt < 2; ++mt) {
#pragma unroll
            for (int v = 0; v < 8; ++v) {
                float r = acc[mt][nt][v] + bv;
                if (relu) r = fmaxf(r, 0.f);
                out[(size_t)(mW + (mt << 4) + hi + v) * ldOut + col] = r;
            }
        }
    }
}

// ---------- WMMA GEMM fused with global-max over points (for g2b) ----------
__global__ __launch_bounds__(128) void k_gemm_gmax(const unsigned short* __restrict__ A,
                                                   const unsigned short* __restrict__ W,
                                                   const float* __restrict__ bias,
                                                   unsigned* __restrict__ gmax, int K, int ldA) {
    int wave = threadIdx.x >> 5;
    int nTile = blockIdx.x << 6;
    int mW = (blockIdx.y << 7) + (wave << 5);
    v8f acc[2][4] = {};
    for (int kk = 0; kk < K; kk += 32) {
        v16bf a0 = load_a_frag(A, ldA, mW, kk);
        v16bf a1 = load_a_frag(A, ldA, mW + 16, kk);
#pragma unroll
        for (int nt = 0; nt < 4; ++nt) {
            v16bf b = load_b_frag(W, K, nTile + (nt << 4), kk);
            acc[0][nt] = __builtin_amdgcn_wmma_f32_16x16x32_bf16(false, a0, false, b,
                                                                 (short)0, acc[0][nt], false, false);
            acc[1][nt] = __builtin_amdgcn_wmma_f32_16x16x32_bf16(false, a1, false, b,
                                                                 (short)0, acc[1][nt], false, false);
        }
    }
    int lane = threadIdx.x & 31;
    int bb = mW / N_;                             // 32 rows never straddle a batch
#pragma unroll
    for (int nt = 0; nt < 4; ++nt) {
        int col = nTile + (nt << 4) + (lane & 15);
        float bv = bias[col];
        float mx = -3.4e38f;
#pragma unroll
        for (int mt = 0; mt < 2; ++mt)
#pragma unroll
            for (int v = 0; v < 8; ++v) mx = fmaxf(mx, acc[mt][nt][v] + bv);
        mx = fmaxf(mx, __shfl_xor(mx, 16, 32));   // merge the two row-halves per column
        if (lane < 16) atomicMax(&gmax[bb * 1024 + col], fkey(mx));
    }
}

// ---------- fused WMMA pairwise-distance + top-16 selection ----------
// 512 threads = 16 waves per block; block owns 16 query rows of one batch.
// Phase 1: A-frags for the 16 rows are preloaded once per wave (compile-time
//          KPAD -> register array), then each wave streams 8 of the 128 column
//          tiles: KPAD/32 B-loads + WMMAs each, neg-distances into a 16x2048
//          f32 LDS panel (128 KB of the WGP's 320 KB).
// Phase 2: wave r runs iterative argmax top-16 on row r via wave32 shuffles.
template <int KPAD>
__global__ __launch_bounds__(512) void k_knn(const unsigned short* __restrict__ X,
                                             const float* __restrict__ norms,
                                             int* __restrict__ idxOut) {
    extern __shared__ float panel[];              // 16 * N_ floats
    constexpr int NK = KPAD / 32;
    int b = blockIdx.y;
    int mLoc = blockIdx.x << 4;
    int mBase = b * N_ + mLoc;
    int wave = threadIdx.x >> 5, lane = threadIdx.x & 31;

    v16bf afr[NK];
#pragma unroll
    for (int t = 0; t < NK; ++t) afr[t] = load_a_frag(X, KPAD, mBase, t << 5);

    for (int ct = wave; ct < (N_ >> 4); ct += 16) {
        int nLoc = ct << 4;
        v8f acc = {};
#pragma unroll
        for (int t = 0; t < NK; ++t) {
            v16bf bf = load_b_frag(X, KPAD, b * N_ + nLoc, t << 5);
            acc = __builtin_amdgcn_wmma_f32_16x16x32_bf16(false, afr[t], false, bf,
                                                          (short)0, acc, false, false);
        }
        int col = nLoc + (lane & 15), hi = (lane >> 4) << 3;
        float xxn = norms[b * N_ + col];
#pragma unroll
        for (int v = 0; v < 8; ++v) {
            int r = hi + v;
            panel[r * N_ + col] = 2.f * acc[v] - norms[mBase + r] - xxn;
        }
    }
    __syncthreads();

    float* rowp = panel + wave * N_;
    int outBase = (mBase + wave) * KNN;
    for (int it = 0; it < KNN; ++it) {
        float best = -3.4e38f; int bi = N_;
        for (int c = lane; c < N_; c += 32) {
            float v = rowp[c];
            if (v > best || (v == best && c < bi)) { best = v; bi = c; }
        }
#pragma unroll
        for (int off = 16; off; off >>= 1) {
            float ov = __shfl_xor(best, off, 32);
            int   oi = __shfl_xor(bi, off, 32);
            if (ov > best || (ov == best && oi < bi)) { best = ov; bi = oi; }
        }
        if (lane == 0) { idxOut[outBase + it] = bi; rowp[bi] = -3.4e38f; }
        asm volatile("s_wait_dscnt 0x0" ::: "memory");  // LDS clear visible before rescan
    }
}

// ---------- final tiny MLP (M = 8): h = relu(W2a g + b); out = W2b h + b ----------
__global__ __launch_bounds__(512) void k_mlp(const unsigned* __restrict__ gmax,
                                             const float* __restrict__ w2a, const float* __restrict__ bw2a,
                                             const float* __restrict__ w2b, const float* __restrict__ bw2b,
                                             float* __restrict__ out) {
    __shared__ float gs[1024];
    __shared__ float hs[512];
    int b = blockIdx.x, t = threadIdx.x;
    for (int c = t; c < 1024; c += 512) gs[c] = fdec(gmax[b * 1024 + c]);
    __syncthreads();
    float s = bw2a[t];
    for (int c = 0; c < 1024; ++c) s += w2a[t * 1024 + c] * gs[c];
    hs[t] = fmaxf(s, 0.f);
    __syncthreads();
    float o = bw2b[t];
    for (int c = 0; c < 512; ++c) o += w2b[t * 512 + c] * hs[c];
    out[b * 512 + t] = o;
}

// ---------- host ----------
extern "C" void kernel_launch(void* const* d_in, const int* in_sizes, int n_in,
                              void* d_out, int out_size, void* d_ws, size_t ws_size,
                              hipStream_t stream) {
    (void)in_sizes; (void)n_in; (void)out_size; (void)ws_size;
    const float* pts  = (const float*)d_in[0];
    const float* w1a  = (const float*)d_in[1];  const float* bw1a = (const float*)d_in[2];
    const float* w1b  = (const float*)d_in[3];  const float* bw1b = (const float*)d_in[4];
    const float* w1c  = (const float*)d_in[5];  const float* bw1c = (const float*)d_in[6];
    const float* g1a  = (const float*)d_in[7];  const float* bg1a = (const float*)d_in[8];
    const float* g1b  = (const float*)d_in[9];  const float* bg1b = (const float*)d_in[10];
    const float* g2a  = (const float*)d_in[11]; const float* bg2a = (const float*)d_in[12];
    const float* g2b  = (const float*)d_in[13]; const float* bg2b = (const float*)d_in[14];
    const float* w2a  = (const float*)d_in[15]; const float* bw2a = (const float*)d_in[16];
    const float* w2b  = (const float*)d_in[17]; const float* bw2b = (const float*)d_in[18];

    char* ws = (char*)d_ws;
    float*          xa    = (float*)(ws);                       // BN*128 f32 (8 MB)
    float*          xb    = (float*)(ws + 8388608);             // BN*128 f32 (8 MB)
    unsigned short* bfA   = (unsigned short*)(ws + 16777216);   // BN*128 bf16 (4 MB)
    unsigned short* wbf   = (unsigned short*)(ws + 20971520);   // 1024*128 bf16
    float*          norms = (float*)(ws + 21233664);            // BN f32
    int*            idx   = (int*)(ws + 21299200);              // BN*16 i32
    unsigned*       gmax  = (unsigned*)(ws + 23396352);         // B*1024 u32

    const size_t lds = (size_t)16 * N_ * 4;                     // 128 KB panel
    dim3 knnGrid(N_ / 16, B_);
    dim3 gemmBlk(128);

    // ---- kNN #1 on raw xyz (Kpad=32)
    k_convert<<<BN * 32 / 256, 256, 0, stream>>>(pts, bfA, 3, 3, 32, BN * 32);
    k_norms  <<<BN / 256, 256, 0, stream>>>(pts, 3, 3, norms);
    k_knn<32><<<knnGrid, 512, lds, stream>>>(bfA, norms, idx);
    k_cov    <<<BN / 256, 256, 0, stream>>>(pts, idx, xa);

    // ---- conv w1a (12->64) relu
    k_convert<<<BN * 32 / 256, 256, 0, stream>>>(xa, bfA, 32, 12, 32, BN * 32);
    k_convert<<<64 * 32 / 256, 256, 0, stream>>>(w1a, wbf, 12, 12, 32, 64 * 32);
    k_gemm<<<dim3(1, BN / 128), gemmBlk, 0, stream>>>(bfA, wbf, bw1a, xb, 32, 32, 64, 1);
    // ---- conv w1b (64->64) relu
    k_convert<<<BN * 64 / 256, 256, 0, stream>>>(xb, bfA, 64, 64, 64, BN * 64);
    k_convert<<<64 * 64 / 256, 256, 0, stream>>>(w1b, wbf, 64, 64, 64, 64 * 64);
    k_gemm<<<dim3(1, BN / 128), gemmBlk, 0, stream>>>(bfA, wbf, bw1b, xa, 64, 64, 64, 1);
    // ---- conv w1c (64->64) relu
    k_convert<<<BN * 64 / 256, 256, 0, stream>>>(xa, bfA, 64, 64, 64, BN * 64);
    k_convert<<<64 * 64 / 256, 256, 0, stream>>>(w1c, wbf, 64, 64, 64, 64 * 64);
    k_gemm<<<dim3(1, BN / 128), gemmBlk, 0, stream>>>(bfA, wbf, bw1c, xb, 64, 64, 64, 1);

    // ---- kNN #2 on 64-ch features, then local max-pool
    k_norms  <<<BN / 256, 256, 0, stream>>>(xb, 64, 64, norms);
    k_convert<<<BN * 64 / 256, 256, 0, stream>>>(xb, bfA, 64, 64, 64, BN * 64);
    k_knn<64><<<knnGrid, 512, lds, stream>>>(bfA, norms, idx);
    k_maxpool<<<BN * 64 / 256, 256, 0, stream>>>(xb, idx, xa, 64, 64);

    // ---- conv g1a (64->64) relu, g1b (64->128)
    k_convert<<<BN * 64 / 256, 256, 0, stream>>>(xa, bfA, 64, 64, 64, BN * 64);
    k_convert<<<64 * 64 / 256, 256, 0, stream>>>(g1a, wbf, 64, 64, 64, 64 * 64);
    k_gemm<<<dim3(1, BN / 128), gemmBlk, 0, stream>>>(bfA, wbf, bg1a, xb, 64, 64, 64, 1);
    k_convert<<<BN * 64 / 256, 256, 0, stream>>>(xb, bfA, 64, 64, 64, BN * 64);
    k_convert<<<128 * 64 / 256, 256, 0, stream>>>(g1b, wbf, 64, 64, 64, 128 * 64);
    k_gemm<<<dim3(2, BN / 128), gemmBlk, 0, stream>>>(bfA, wbf, bg1b, xa, 64, 64, 128, 0);

    // ---- kNN #3 on 128-ch features, then local max-pool
    k_norms  <<<BN / 256, 256, 0, stream>>>(xa, 128, 128, norms);
    k_convert<<<BN * 128 / 256, 256, 0, stream>>>(xa, bfA, 128, 128, 128, BN * 128);
    k_knn<128><<<knnGrid, 512, lds, stream>>>(bfA, norms, idx);
    k_maxpool<<<BN * 128 / 256, 256, 0, stream>>>(xa, idx, xb, 128, 128);

    // ---- conv g2a (128->128) relu
    k_convert<<<BN * 128 / 256, 256, 0, stream>>>(xb, bfA, 128, 128, 128, BN * 128);
    k_convert<<<128 * 128 / 256, 256, 0, stream>>>(g2a, wbf, 128, 128, 128, 128 * 128);
    k_gemm<<<dim3(2, BN / 128), gemmBlk, 0, stream>>>(bfA, wbf, bg2a, xa, 128, 128, 128, 1);

    // ---- conv g2b (128->1024) fused with global max over all points
    k_convert<<<BN * 128 / 256, 256, 0, stream>>>(xa, bfA, 128, 128, 128, BN * 128);
    k_convert<<<1024 * 128 / 256, 256, 0, stream>>>(g2b, wbf, 128, 128, 128, 1024 * 128);
    k_zero<<<(B_ * 1024) / 256, 256, 0, stream>>>(gmax, B_ * 1024);
    k_gemm_gmax<<<dim3(16, BN / 128), gemmBlk, 0, stream>>>(bfA, wbf, bg2b, gmax, 128, 128);

    // ---- final MLP (1024 -> 512 relu -> 512)
    k_mlp<<<dim3(B_), 512, 0, stream>>>(gmax, w2a, bw2a, w2b, bw2b, (float*)d_out);
}
